// ExpertParallelMoE_53231824666819
// MI455X (gfx1250) — matrix operand order repaired and checked by reading, not compiled, not run on hardware
//
#include <hip/hip_runtime.h>
#include <math.h>

// ---------------------------------------------------------------------------
// MoE top-2 SwiGLU, sparse grouped-GEMM implementation for gfx1250 (wave32).
// Matrix cores: v_wmma_f32_16x16x32_bf16 (fp32 accumulate).
// Double-buffered LDS (1 barrier / K-step), packed bf16 LDS stores,
// global_prefetch on the weight/activation streams.
// ---------------------------------------------------------------------------

typedef __bf16 bf16_t;
typedef __bf16  v16bf __attribute__((ext_vector_type(16)));
typedef __bf16  v4bf  __attribute__((ext_vector_type(4)));
typedef __bf16  v2bf  __attribute__((ext_vector_type(2)));
typedef float   v8f   __attribute__((ext_vector_type(8)));

#define NTOK 8192      // B*S tokens
#define DDIM 2048
#define FDIM 1408
#define NEXP 8
#define MT   128       // token tile (M)
#define FT   64        // f tile (pass1 N-dim)
#define DT   64        // d tile (pass2 N-dim)
#define KT   32        // WMMA K per step (bf16)
#define LDSS 40        // LDS row stride in bf16 (32 + 8 pad, 80B = 16B aligned)

// ---- WMMA helpers ---------------------------------------------------------

static __device__ __forceinline__ v8f wmma_bf16(v16bf a, v16bf b, v8f c) {
  return __builtin_amdgcn_wmma_f32_16x16x32_bf16(false, a, false, b,
                                                 (short)0, c, false, false);
}

static __device__ __forceinline__ v8f v8f_zero() {
  v8f z;
#pragma unroll
  for (int i = 0; i < 8; ++i) z[i] = 0.0f;
  return z;
}

// A fragment (16x32 bf16): lane&15 = M row; lane>>4 selects K-half.
// VGPR 0..3 <- K = khalf*8 + 0..7 (16B), VGPR 4..7 <- K = 16 + khalf*8 + 0..7.
static __device__ __forceinline__ v16bf lds_frag_a(const bf16_t* tile, int row,
                                                   int khalf) {
  union { v16bf v; uint4 q[2]; } u;
  const bf16_t* p = tile + row * LDSS;
  u.q[0] = *(const uint4*)(p + khalf * 8);
  u.q[1] = *(const uint4*)(p + 16 + khalf * 8);
  return u.v;
}

// B fragment (32x16 bf16), LDS tile stored transposed [N][K]:
// lane&15 = N col; VGPR 0..3 <- K = khalf*8 + 0..7, VGPR 4..7 <- K = 16 + khalf*8 + 0..7.
static __device__ __forceinline__ v16bf lds_frag_b(const bf16_t* tile, int col,
                                                   int khalf) {
  union { v16bf v; uint4 q[2]; } u;
  const bf16_t* p = tile + col * LDSS;
  u.q[0] = *(const uint4*)(p + khalf * 8);
  u.q[1] = *(const uint4*)(p + 16 + khalf * 8);
  return u.v;
}

// Pack two fp32 K-neighbors into one bf16x2 LDS word (single ds_store_b32).
static __device__ __forceinline__ void st_pair(bf16_t* dst, float a, float b) {
  v2bf pr;
  pr[0] = (bf16_t)a;
  pr[1] = (bf16_t)b;
  *(v2bf*)dst = pr;
}

// Fast SiLU: g * 1/(1+exp(-g)) with v_rcp_f32 instead of IEEE divide.
static __device__ __forceinline__ float fast_silu(float g) {
  return g * __builtin_amdgcn_rcpf(1.0f + __expf(-g));
}

// ---- Kernel 0: zero output + counters -------------------------------------

__global__ void k_zero(float* out, size_t n, int* counts, float* psum) {
  size_t i = (size_t)blockIdx.x * blockDim.x + threadIdx.x;
  size_t stride = (size_t)gridDim.x * blockDim.x;
  for (; i < n; i += stride) out[i] = 0.0f;
  if (blockIdx.x == 0 && threadIdx.x < NEXP) {
    counts[threadIdx.x] = 0;
    psum[threadIdx.x] = 0.0f;
  }
}

// ---- Kernel 1: router (one wave32 per token) ------------------------------

__global__ void k_router(const float* __restrict__ x,
                         const float* __restrict__ rw, int* counts, float* psum,
                         int* topi, float* topw) {
  const int lane = threadIdx.x & 31;
  const int wave = threadIdx.x >> 5;
  const int t = blockIdx.x * 8 + wave;
  float acc[NEXP];
#pragma unroll
  for (int e = 0; e < NEXP; ++e) acc[e] = 0.0f;
  const float* xr = x + (size_t)t * DDIM;
  for (int i = 0; i < DDIM / 32; ++i) {
    const int d = i * 32 + lane;
    const float xv = xr[d];
    const float4* rp = (const float4*)(rw + d * NEXP);
    const float4 r0 = rp[0], r1 = rp[1];
    acc[0] += xv * r0.x; acc[1] += xv * r0.y;
    acc[2] += xv * r0.z; acc[3] += xv * r0.w;
    acc[4] += xv * r1.x; acc[5] += xv * r1.y;
    acc[6] += xv * r1.z; acc[7] += xv * r1.w;
  }
#pragma unroll
  for (int off = 16; off > 0; off >>= 1) {
#pragma unroll
    for (int e = 0; e < NEXP; ++e) acc[e] += __shfl_xor(acc[e], off, 32);
  }
  if (lane == 0) {
    float m = acc[0];
#pragma unroll
    for (int e = 1; e < NEXP; ++e) m = fmaxf(m, acc[e]);
    float p[NEXP], s = 0.0f;
#pragma unroll
    for (int e = 0; e < NEXP; ++e) { p[e] = __expf(acc[e] - m); s += p[e]; }
    const float inv = __builtin_amdgcn_rcpf(s);
#pragma unroll
    for (int e = 0; e < NEXP; ++e) p[e] *= inv;
    int i0 = 0;
#pragma unroll
    for (int e = 1; e < NEXP; ++e) if (p[e] > p[i0]) i0 = e;
    int i1 = (i0 == 0) ? 1 : 0;
#pragma unroll
    for (int e = 0; e < NEXP; ++e) if (e != i0 && p[e] > p[i1]) i1 = e;
    const float winv = __builtin_amdgcn_rcpf(p[i0] + p[i1]);
    topi[2 * t] = i0;  topi[2 * t + 1] = i1;
    topw[2 * t] = p[i0] * winv;  topw[2 * t + 1] = p[i1] * winv;
    atomicAdd(&counts[i0], 1);
    atomicAdd(&counts[i1], 1);
#pragma unroll
    for (int e = 0; e < NEXP; ++e) atomicAdd(&psum[e], p[e]);
  }
}

// ---- Kernel 2: prefix scan of counts + aux loss ---------------------------

__global__ void k_scan(const int* counts, const float* psum, int* basep,
                       int* cursor, float* aux_out) {
  if (threadIdx.x == 0 && blockIdx.x == 0) {
    int b = 0;
    float aux = 0.0f;
#pragma unroll
    for (int e = 0; e < NEXP; ++e) {
      basep[e] = b;
      cursor[e] = b;
      const float f = (float)counts[e] / (float)NTOK;
      const float P = psum[e] / (float)NTOK;
      aux += f * P;
      b += counts[e];
    }
    aux_out[0] = aux * ((float)NEXP / 2.0f);   // E / TOP_K
  }
}

// ---- Kernel 3: bucketize assignments into a flat compact list -------------

__global__ void k_bucket(const int* __restrict__ topi,
                         const float* __restrict__ topw, int* cursor,
                         int* buckets, float* gatew) {
  const int t = blockIdx.x * blockDim.x + threadIdx.x;
  if (t >= NTOK) return;
#pragma unroll
  for (int k = 0; k < 2; ++k) {
    const int e = topi[2 * t + k];
    const int pos = atomicAdd(&cursor[e], 1);
    buckets[pos] = t;
    gatew[pos] = topw[2 * t + k];
  }
}

// ---- Kernel 4: gate/up GEMMs + SiLU (H = silu(x Wg) * (x Wu), bf16) -------

__global__ void __launch_bounds__(256) k_ffn_gate_up(
    const float* __restrict__ x, const float* __restrict__ w_gate,
    const float* __restrict__ w_up, const int* __restrict__ counts,
    const int* __restrict__ basep, const int* __restrict__ buckets,
    bf16_t* __restrict__ H) {
  const int e = blockIdx.z;
  const int mt = blockIdx.y;
  const int cnt = counts[e];
  if (mt * MT >= cnt) return;
  const int rows = min(MT, cnt - mt * MT);
  const int b0 = basep[e] + mt * MT;
  const int f0 = blockIdx.x * FT;

  __shared__ __align__(16) bf16_t aT[2][MT * LDSS];
  __shared__ __align__(16) bf16_t bG[2][FT * LDSS];
  __shared__ __align__(16) bf16_t bU[2][FT * LDSS];
  __shared__ int tokL[MT];

  const int tid = threadIdx.x;
  if (tid < MT) tokL[tid] = buckets[b0 + min(tid, rows - 1)];
  __syncthreads();

  const float* wg = w_gate + (size_t)e * DDIM * FDIM;
  const float* wu = w_up + (size_t)e * DDIM * FDIM;

  const int lane = tid & 31, wv = tid >> 5;
  const int mbase = (wv & 3) * 32;     // 4 waves cover 128 M
  const int nbase = (wv >> 2) * 32;    // 2 waves cover 64 N
  const int col = lane & 15, half = lane >> 4;

  // B-tile loader mapping: each thread owns 2 consecutive K rows x 4 f cols.
  const int bkr = (tid >> 4) * 2;        // 0,2,..,30
  const int bfg = (tid & 15) * 4;        // 0,4,..,60

  v8f accg[2][2], accu[2][2];
#pragma unroll
  for (int i = 0; i < 2; ++i)
#pragma unroll
    for (int j = 0; j < 2; ++j) { accg[i][j] = v8f_zero(); accu[i][j] = v8f_zero(); }

  const int KSTEPS = DDIM / KT;
  for (int kk = 0; kk < KSTEPS; ++kk) {
    const int k0 = kk * KT;
    const int p = kk & 1;
    // A tile: gather 128 token rows x 32 K (fp32 -> bf16, packed b64 stores)
#pragma unroll
    for (int rep = 0; rep < 4; ++rep) {
      const int idx = rep * 256 + tid;
      const int row = idx >> 3, cg = idx & 7;
      const float* src = x + (size_t)tokL[row] * DDIM + k0 + cg * 4;
      const float4 xv = *(const float4*)src;
      if (kk + 1 < KSTEPS) __builtin_prefetch((const void*)(src + KT), 0, 3);
      v4bf pk;
      pk[0] = (bf16_t)xv.x; pk[1] = (bf16_t)xv.y;
      pk[2] = (bf16_t)xv.z; pk[3] = (bf16_t)xv.w;
      *(v4bf*)(&aT[p][row * LDSS + cg * 4]) = pk;
    }
    // B tiles: 32 K rows x 64 F cols, transposed into LDS as [F][K].
    // Two adjacent K rows per thread -> packed bf16x2 (ds_store_b32) writes.
    {
      const size_t g0 = (size_t)(k0 + bkr) * FDIM + f0 + bfg;
      const float4 gv0 = *(const float4*)(wg + g0);
      const float4 gv1 = *(const float4*)(wg + g0 + FDIM);
      const float4 uv0 = *(const float4*)(wu + g0);
      const float4 uv1 = *(const float4*)(wu + g0 + FDIM);
      if (kk + 1 < KSTEPS) {
        __builtin_prefetch((const void*)(wg + g0 + (size_t)KT * FDIM), 0, 3);
        __builtin_prefetch((const void*)(wu + g0 + (size_t)KT * FDIM), 0, 3);
      }
      st_pair(&bG[p][(bfg + 0) * LDSS + bkr], gv0.x, gv1.x);
      st_pair(&bG[p][(bfg + 1) * LDSS + bkr], gv0.y, gv1.y);
      st_pair(&bG[p][(bfg + 2) * LDSS + bkr], gv0.z, gv1.z);
      st_pair(&bG[p][(bfg + 3) * LDSS + bkr], gv0.w, gv1.w);
      st_pair(&bU[p][(bfg + 0) * LDSS + bkr], uv0.x, uv1.x);
      st_pair(&bU[p][(bfg + 1) * LDSS + bkr], uv0.y, uv1.y);
      st_pair(&bU[p][(bfg + 2) * LDSS + bkr], uv0.z, uv1.z);
      st_pair(&bU[p][(bfg + 3) * LDSS + bkr], uv0.w, uv1.w);
    }
    __syncthreads();   // lowers to s_wait_dscnt 0 + barrier: fences prior reads

    v16bf a[2], bg[2], bu[2];
#pragma unroll
    for (int mi = 0; mi < 2; ++mi)
      a[mi] = lds_frag_a(aT[p], mbase + mi * 16 + col, half);
#pragma unroll
    for (int ni = 0; ni < 2; ++ni) {
      bg[ni] = lds_frag_b(bG[p], nbase + ni * 16 + col, half);
      bu[ni] = lds_frag_b(bU[p], nbase + ni * 16 + col, half);
    }
#pragma unroll
    for (int mi = 0; mi < 2; ++mi)
#pragma unroll
      for (int ni = 0; ni < 2; ++ni) {
        accg[mi][ni] = wmma_bf16(a[mi], bg[ni], accg[mi][ni]);
        accu[mi][ni] = wmma_bf16(a[mi], bu[ni], accu[mi][ni]);
      }
  }

  // Epilogue: h = silu(g) * u, store bf16 into compact H.
  if (rows == MT) {
#pragma unroll
    for (int mi = 0; mi < 2; ++mi)
#pragma unroll
      for (int ni = 0; ni < 2; ++ni) {
        const v8f g = accg[mi][ni], u = accu[mi][ni];
#pragma unroll
        for (int r = 0; r < 8; ++r) {
          const int rl = mbase + mi * 16 + half * 8 + r;
          const float h = fast_silu(g[r]) * u[r];
          H[(size_t)(b0 + rl) * FDIM + f0 + nbase + ni * 16 + col] = (bf16_t)h;
        }
      }
  } else {
#pragma unroll
    for (int mi = 0; mi < 2; ++mi)
#pragma unroll
      for (int ni = 0; ni < 2; ++ni) {
        const v8f g = accg[mi][ni], u = accu[mi][ni];
#pragma unroll
        for (int r = 0; r < 8; ++r) {
          const int rl = mbase + mi * 16 + half * 8 + r;
          if (rl < rows) {
            const float h = fast_silu(g[r]) * u[r];
            H[(size_t)(b0 + rl) * FDIM + f0 + nbase + ni * 16 + col] = (bf16_t)h;
          }
        }
      }
  }
}

// ---- Kernel 5: down GEMM + gated scatter-add into output ------------------

__global__ void __launch_bounds__(256) k_ffn_down(
    const bf16_t* __restrict__ H, const float* __restrict__ w_down,
    const int* __restrict__ counts, const int* __restrict__ basep,
    const int* __restrict__ buckets, const float* __restrict__ gatew,
    float* __restrict__ out) {
  const int e = blockIdx.z;
  const int mt = blockIdx.y;
  const int cnt = counts[e];
  if (mt * MT >= cnt) return;
  const int rows = min(MT, cnt - mt * MT);
  const int b0 = basep[e] + mt * MT;
  const int d0 = blockIdx.x * DT;

  __shared__ __align__(16) bf16_t aT[2][MT * LDSS];
  __shared__ __align__(16) bf16_t bD[2][DT * LDSS];
  __shared__ int tokL[MT];
  __shared__ float wL[MT];

  const int tid = threadIdx.x;
  if (tid < MT) {
    const int slot = b0 + min(tid, rows - 1);
    tokL[tid] = buckets[slot];
    wL[tid] = gatew[slot];
  }
  __syncthreads();

  const float* wd = w_down + (size_t)e * FDIM * DDIM;

  const int lane = tid & 31, wv = tid >> 5;
  const int mbase = (wv & 3) * 32;
  const int nbase = (wv >> 2) * 32;
  const int col = lane & 15, half = lane >> 4;

  const int bkr = (tid >> 4) * 2;
  const int bdg = (tid & 15) * 4;

  v8f acc[2][2];
#pragma unroll
  for (int i = 0; i < 2; ++i)
#pragma unroll
    for (int j = 0; j < 2; ++j) acc[i][j] = v8f_zero();

  const int KSTEPS = FDIM / KT;
  for (int kk = 0; kk < KSTEPS; ++kk) {
    const int k0 = kk * KT;
    const int p = kk & 1;
    // A tile: 128 rows x 32 K of H (already bf16), 16B vector copies
#pragma unroll
    for (int rep = 0; rep < 2; ++rep) {
      const int idx = rep * 256 + tid;
      const int row = idx >> 2, grp = idx & 3;
      const int slot = b0 + min(row, rows - 1);
      const bf16_t* src = H + (size_t)slot * FDIM + k0 + grp * 8;
      if (kk + 1 < KSTEPS) __builtin_prefetch((const void*)(src + KT), 0, 3);
      *(uint4*)(&aT[p][row * LDSS + grp * 8]) = *(const uint4*)src;
    }
    // B tile: 32 K (f) rows x 64 D cols, transposed into LDS as [D][K].
    {
      const float* src = wd + (size_t)(k0 + bkr) * DDIM + d0 + bdg;
      const float4 dv0 = *(const float4*)src;
      const float4 dv1 = *(const float4*)(src + DDIM);
      if (kk + 1 < KSTEPS)
        __builtin_prefetch((const void*)(src + (size_t)KT * DDIM), 0, 3);
      st_pair(&bD[p][(bdg + 0) * LDSS + bkr], dv0.x, dv1.x);
      st_pair(&bD[p][(bdg + 1) * LDSS + bkr], dv0.y, dv1.y);
      st_pair(&bD[p][(bdg + 2) * LDSS + bkr], dv0.z, dv1.z);
      st_pair(&bD[p][(bdg + 3) * LDSS + bkr], dv0.w, dv1.w);
    }
    __syncthreads();

    v16bf a[2], b[2];
#pragma unroll
    for (int mi = 0; mi < 2; ++mi)
      a[mi] = lds_frag_a(aT[p], mbase + mi * 16 + col, half);
#pragma unroll
    for (int ni = 0; ni < 2; ++ni)
      b[ni] = lds_frag_b(bD[p], nbase + ni * 16 + col, half);
#pragma unroll
    for (int mi = 0; mi < 2; ++mi)
#pragma unroll
      for (int ni = 0; ni < 2; ++ni)
        acc[mi][ni] = wmma_bf16(a[mi], b[ni], acc[mi][ni]);
  }

  // Epilogue: gate-weighted atomic accumulate into out[token, d]
  if (rows == MT) {
#pragma unroll
    for (int mi = 0; mi < 2; ++mi)
#pragma unroll
      for (int ni = 0; ni < 2; ++ni) {
        const v8f c = acc[mi][ni];
#pragma unroll
        for (int r = 0; r < 8; ++r) {
          const int rl = mbase + mi * 16 + half * 8 + r;
          atomicAdd(&out[(size_t)tokL[rl] * DDIM + d0 + nbase + ni * 16 + col],
                    wL[rl] * c[r]);
        }
      }
  } else {
#pragma unroll
    for (int mi = 0; mi < 2; ++mi)
#pragma unroll
      for (int ni = 0; ni < 2; ++ni) {
        const v8f c = acc[mi][ni];
#pragma unroll
        for (int r = 0; r < 8; ++r) {
          const int rl = mbase + mi * 16 + half * 8 + r;
          if (rl < rows) {
            atomicAdd(
                &out[(size_t)tokL[rl] * DDIM + d0 + nbase + ni * 16 + col],
                wL[rl] * c[r]);
          }
        }
      }
  }
}

// ---- Host-side launcher ---------------------------------------------------

extern "C" void kernel_launch(void* const* d_in, const int* in_sizes, int n_in,
                              void* d_out, int out_size, void* d_ws,
                              size_t ws_size, hipStream_t stream) {
  const float* x  = (const float*)d_in[0];   // [N, D]
  const float* rw = (const float*)d_in[1];   // [D, E]
  const float* wg = (const float*)d_in[2];   // [E, D, F]
  const float* wu = (const float*)d_in[3];   // [E, D, F]
  const float* wd = (const float*)d_in[4];   // [E, F, D]
  float* out = (float*)d_out;                // [N*D] out ++ [1] aux

  char* ws = (char*)d_ws;
  int*    counts  = (int*)(ws + 0);               // 8 ints
  int*    cursor  = (int*)(ws + 32);              // 8 ints
  int*    basep   = (int*)(ws + 64);              // 8 ints
  float*  psum    = (float*)(ws + 96);            // 8 floats
  int*    topi    = (int*)(ws + 256);             // 2N ints
  float*  topw    = (float*)(ws + 256 + 65536);   // 2N floats
  int*    buckets = (int*)(ws + 131328);          // 2N ints
  float*  gatew   = (float*)(ws + 196864);        // 2N floats
  bf16_t* H       = (bf16_t*)(ws + 262400);       // 2N * F bf16 (~46 MB)

  const size_t outN = (size_t)NTOK * DDIM;

  k_zero<<<2048, 256, 0, stream>>>(out, outN + 1, counts, psum);
  k_router<<<NTOK / 8, 256, 0, stream>>>(x, rw, counts, psum, topi, topw);
  k_scan<<<1, 32, 0, stream>>>(counts, psum, basep, cursor, out + outN);
  k_bucket<<<NTOK / 256, 256, 0, stream>>>(topi, topw, cursor, buckets, gatew);

  dim3 g1(FDIM / FT, (NTOK + MT - 1) / MT, NEXP);   // 22 x 64 x 8
  k_ffn_gate_up<<<g1, 256, 0, stream>>>(x, wg, wu, counts, basep, buckets, H);

  dim3 g2(DDIM / DT, (NTOK + MT - 1) / MT, NEXP);   // 32 x 64 x 8
  k_ffn_down<<<g2, 256, 0, stream>>>(H, wd, counts, basep, buckets, gatew, out);
}
